// PoseNet_21663815041430
// MI455X (gfx1250) — matrix operand order repaired and checked by reading, not compiled, not run on hardware
//
#include <hip/hip_runtime.h>
#include <hip/hip_bf16.h>
#include <math.h>

typedef __attribute__((ext_vector_type(16))) _Float16 v16h;
typedef __attribute__((ext_vector_type(8)))  _Float16 v8h;
typedef __attribute__((ext_vector_type(8)))  float    v8f;

#define N_PTS 2048
#define KNN_K 16
#define OBJ_N 21
#define ROT_R 12
#define HW_SZ 6400
#define PSPC  32

// ---------------------------------------------------------------------------
// KNN: per-point 16 smallest squared distances, stable unrolled insertion sort
// ---------------------------------------------------------------------------
__global__ void knn_kernel(const float* __restrict__ cloud, int* __restrict__ idx, int Nn) {
    int i = blockIdx.x * blockDim.x + threadIdx.x;
    if (i >= Nn) return;
    float px = cloud[i * 3 + 0], py = cloud[i * 3 + 1], pz = cloud[i * 3 + 2];
    float bd[KNN_K]; int bi[KNN_K];
#pragma unroll
    for (int s = 0; s < KNN_K; ++s) { bd[s] = 3.0e38f; bi[s] = 0; }
    for (int j = 0; j < Nn; ++j) {
        float dx = cloud[j * 3 + 0] - px;
        float dy = cloud[j * 3 + 1] - py;
        float dz = cloud[j * 3 + 2] - pz;
        float d = dx * dx + dy * dy + dz * dz;
        int ji = j;
#pragma unroll
        for (int s = 0; s < KNN_K; ++s) {
            bool sw = d < bd[s];
            float td = sw ? bd[s] : d;  bd[s] = sw ? d : bd[s];  d = td;
            int   ti = sw ? bi[s] : ji; bi[s] = sw ? ji : bi[s]; ji = ti;
        }
    }
#pragma unroll
    for (int s = 0; s < KNN_K; ++s) idx[i * KNN_K + s] = bi[s];
}

// Zero-pad edge_conv1 weights (64,6) -> (64,32)
__global__ void pad_w1_kernel(const float* __restrict__ W, float* __restrict__ Wp) {
    int i = blockIdx.x * blockDim.x + threadIdx.x;
    if (i < 64 * 32) {
        int o = i >> 5, k = i & 31;
        Wp[i] = (k < 6) ? W[o * 6 + k] : 0.0f;
    }
}

// Gather psp_feat[0] at choice -> y (N, 32) f16
__global__ void gather_kernel(const float* __restrict__ psp, const int* __restrict__ choice,
                              _Float16* __restrict__ y, int Nn) {
    int i = blockIdx.x * blockDim.x + threadIdx.x;
    if (i < Nn * PSPC) {
        int n = i >> 5, c = i & 31;
        y[i] = (_Float16)psp[c * HW_SZ + choice[n]];
    }
}

// Mean over N of X (N, C) f16 -> out (C) f16
__global__ void meanpool_kernel(const _Float16* __restrict__ X, _Float16* __restrict__ out,
                                int C, int Nn) {
    int c = blockIdx.x * blockDim.x + threadIdx.x;
    if (c >= C) return;
    float s = 0.0f;
    for (int n = 0; n < Nn; ++n) s += (float)X[(size_t)n * C + c];
    out[c] = (_Float16)(s / (float)Nn);
}

// Broadcast pooled (1024) into tcat channels [128,1152) for every point
__global__ void bcast_kernel(const _Float16* __restrict__ v, _Float16* __restrict__ tcat, int Nn) {
    int i = blockIdx.x * blockDim.x + threadIdx.x;
    if (i < Nn * 1024) {
        int n = i >> 10, c = i & 1023;
        tcat[(size_t)n * 1152 + 128 + c] = v[c];
    }
}

// ---------------------------------------------------------------------------
// Edge conv: out[n,o] = max_k relu( W[o,:] . [x(n,:) ; x(nb,:)-x(n,:)] + b[o] )
// One wave per (16-row o-tile, point n); WMMA columns are the 16 neighbors.
// All loads unconditional (clamped indices); selects on values only.
// ---------------------------------------------------------------------------
template <typename InT>
__global__ void edgeconv_kernel(const float* __restrict__ W, const float* __restrict__ bias,
                                const InT* __restrict__ X, const int* __restrict__ idx,
                                _Float16* __restrict__ Y,
                                int O, int C, int Nn, int ldX, int ldY, int wstride) {
    const int lane = threadIdx.x & 31;
    const int wave = threadIdx.x >> 5;
    const int o0 = blockIdx.x * 16;
    const int n  = blockIdx.y * 4 + wave;
    const int half16 = lane >> 4;
    const int lane16 = lane & 15;
    const int twoC = 2 * C;

    const int nb = idx[n * KNN_K + lane16];
    const InT* Xc = X + (size_t)n  * ldX;
    const InT* Xn = X + (size_t)nb * ldX;

    v8f acc = {};
    const int ksteps = wstride >> 5;
    for (int s = 0; s < ksteps; ++s) {
        const int c0 = s * 32;
        // A tile: rows always valid (O multiple of 16), wstride multiple of 32
        const float* wrow = W + (size_t)(o0 + lane16) * wstride + c0 + 16 * half16;
        float4 wa = *(const float4*)(wrow + 0);
        float4 wb = *(const float4*)(wrow + 4);
        float4 wc = *(const float4*)(wrow + 8);
        float4 wd = *(const float4*)(wrow + 12);
        v16h a;
        a[0] = (_Float16)wa.x; a[1] = (_Float16)wa.y; a[2]  = (_Float16)wa.z; a[3]  = (_Float16)wa.w;
        a[4] = (_Float16)wb.x; a[5] = (_Float16)wb.y; a[6]  = (_Float16)wb.z; a[7]  = (_Float16)wb.w;
        a[8] = (_Float16)wc.x; a[9] = (_Float16)wc.y; a[10] = (_Float16)wc.z; a[11] = (_Float16)wc.w;
        a[12] = (_Float16)wd.x; a[13] = (_Float16)wd.y; a[14] = (_Float16)wd.z; a[15] = (_Float16)wd.w;

        // B tile: edge feature, unconditional clamped loads + value selects
        v16h b;
#pragma unroll
        for (int h = 0; h < 16; ++h) {
            int ec = c0 + 16 * half16 + h;
            int t_ = (ec < C) ? ec : (ec - C);
            int ci = (t_ < C) ? t_ : (C - 1);
            float vc = (float)Xc[ci];
            float vn = (float)Xn[ci];
            float v = (ec < C) ? vc : ((ec < twoC) ? (vn - vc) : 0.0f);
            b[h] = (_Float16)v;
        }
        acc = __builtin_amdgcn_wmma_f32_16x16x32_f16(false, a, false, b, (short)0, acc, false, false);
    }

    // bias + relu + max over the 16 neighbor columns (per 16-lane half)
    float vmax[8];
#pragma unroll
    for (int j = 0; j < 8; ++j) vmax[j] = fmaxf(acc[j] + bias[o0 + 8 * half16 + j], 0.0f);
#pragma unroll
    for (int off = 1; off < 16; off <<= 1) {
#pragma unroll
        for (int j = 0; j < 8; ++j)
            vmax[j] = fmaxf(vmax[j], __shfl_xor(vmax[j], off, 32));
    }
    if (lane16 == 0) {
        _Float16* yp = Y + (size_t)n * ldY + o0 + 8 * half16;
        v8h hv;
#pragma unroll
        for (int j = 0; j < 8; ++j) hv[j] = (_Float16)vmax[j];
        *(v8h*)yp = hv;
    }
}

// ---------------------------------------------------------------------------
// Generic 1x1-conv GEMM: Y[n,o] = act( W[o,:]·X[n,:] + b[o] ), f16 activations.
// Block = 4 waves; each wave: 16(o) x 64(n) tile (4 WMMAs sharing A).
// Cin multiple of 32. act: 0 none, 1 relu, 2 sigmoid.
// ---------------------------------------------------------------------------
template <typename OutT>
__global__ void gemm_act_kernel(const float* __restrict__ W, const float* __restrict__ bias,
                                const _Float16* __restrict__ X, OutT* __restrict__ Y,
                                int O, int Cin, int Nn, int ldX, int ldY, int act) {
    const int lane = threadIdx.x & 31;
    const int wave = threadIdx.x >> 5;
    const int o0 = blockIdx.x * 16;
    const int n0 = blockIdx.y * 256 + wave * 64;
    const int half16 = lane >> 4;
    const int lane16 = lane & 15;
    const int mrow = (o0 + lane16 < O) ? (o0 + lane16) : (O - 1);   // clamped (v_min)

    v8f acc[4] = {};

    for (int c0 = 0; c0 < Cin; c0 += 32) {
        const float* wrow = W + (size_t)mrow * Cin + c0 + 16 * half16;
        float4 wa = *(const float4*)(wrow + 0);
        float4 wb = *(const float4*)(wrow + 4);
        float4 wc = *(const float4*)(wrow + 8);
        float4 wd = *(const float4*)(wrow + 12);
        v16h a;
        a[0] = (_Float16)wa.x; a[1] = (_Float16)wa.y; a[2]  = (_Float16)wa.z; a[3]  = (_Float16)wa.w;
        a[4] = (_Float16)wb.x; a[5] = (_Float16)wb.y; a[6]  = (_Float16)wb.z; a[7]  = (_Float16)wb.w;
        a[8] = (_Float16)wc.x; a[9] = (_Float16)wc.y; a[10] = (_Float16)wc.z; a[11] = (_Float16)wc.w;
        a[12] = (_Float16)wd.x; a[13] = (_Float16)wd.y; a[14] = (_Float16)wd.z; a[15] = (_Float16)wd.w;

        const int cb = c0 + 16 * half16;
#pragma unroll
        for (int t = 0; t < 4; ++t) {
            int col = n0 + t * 16 + lane16;
            col = (col < Nn) ? col : (Nn - 1);                       // clamped (v_min)
            v16h b = *(const v16h*)(X + (size_t)col * ldX + cb);     // 16 contiguous halves
            acc[t] = __builtin_amdgcn_wmma_f32_16x16x32_f16(false, a, false, b, (short)0, acc[t], false, false);
        }
    }

    constexpr int amask = (int)(16 / sizeof(OutT)) - 1;   // alignment mask for packed store
    const bool vecpath = (o0 + 16 <= O) && ((ldY & amask) == 0);
#pragma unroll
    for (int t = 0; t < 4; ++t) {
        const int col = n0 + t * 16 + lane16;
        if (col >= Nn) continue;
        const int ob = o0 + 8 * half16;
        float v[8];
#pragma unroll
        for (int j = 0; j < 8; ++j) {
            int bb = (ob + j < O) ? (ob + j) : (O - 1);
            float z = acc[t][j] + bias[bb];
            if (act == 1)      z = fmaxf(z, 0.0f);
            else if (act == 2) z = 1.0f / (1.0f + __expf(-z));
            v[j] = z;
        }
        OutT* yp = Y + (size_t)col * ldY + ob;
        if (vecpath) {
            if constexpr (sizeof(OutT) == 2) {
                v8h hv;
#pragma unroll
                for (int j = 0; j < 8; ++j) hv[j] = (_Float16)v[j];
                *(v8h*)yp = hv;
            } else {
                *(float4*)(yp + 0) = make_float4(v[0], v[1], v[2], v[3]);
                *(float4*)(yp + 4) = make_float4(v[4], v[5], v[6], v[7]);
            }
        } else {
#pragma unroll
            for (int j = 0; j < 8; ++j) if (ob + j < O) yp[j] = (OutT)v[j];
        }
    }
}

// ---------------------------------------------------------------------------
// Final head. out layout: [out_rx 12*4][out_tx N*3][out_cx 12]
// ---------------------------------------------------------------------------
__global__ void head_kernel(const float* __restrict__ t6, const float* __restrict__ c4,
                            const float* __restrict__ rr4, const int* __restrict__ obj_idx,
                            const float* __restrict__ anchors, float* __restrict__ out, int Nn) {
    const int sel = obj_idx[0];
    const int tid = threadIdx.x;
    if (tid < ROT_R) {
        const int r = tid;
        float q0 = rr4[(size_t)(sel * ROT_R + r) * 4 + 0];
        float q1 = rr4[(size_t)(sel * ROT_R + r) * 4 + 1];
        float q2 = rr4[(size_t)(sel * ROT_R + r) * 4 + 2];
        float q3 = rr4[(size_t)(sel * ROT_R + r) * 4 + 3];
        float nrm = sqrtf(q0 * q0 + q1 * q1 + q2 * q2 + q3 * q3);
        float inv = 1.0f / fmaxf(nrm, 1e-12f);
        float w = q0 * inv, x = q1 * inv, y = q2 * inv, z = q3 * inv;
        float a0 = anchors[r * 4 + 0], a1 = anchors[r * 4 + 1];
        float a2 = anchors[r * 4 + 2], a3 = anchors[r * 4 + 3];
        out[r * 4 + 0] = w * a0 - x * a1 - y * a2 - z * a3;
        out[r * 4 + 1] = x * a0 + w * a1 + z * a2 - y * a3;
        out[r * 4 + 2] = y * a0 - z * a1 + w * a2 + x * a3;
        out[r * 4 + 3] = z * a0 + y * a1 - x * a2 + w * a3;
        out[48 + Nn * 3 + r] = c4[sel * ROT_R + r];
    }
    for (int i = tid; i < Nn * 3; i += blockDim.x) {
        int n = i / 3, j = i % 3;
        out[48 + i] = t6[(size_t)n * 63 + sel * 3 + j];
    }
}

// ---------------------------------------------------------------------------
extern "C" void kernel_launch(void* const* d_in, const int* in_sizes, int n_in,
                              void* d_out, int out_size, void* d_ws, size_t ws_size,
                              hipStream_t stream) {
    const int Nn = N_PTS;
    const float* cloud  = (const float*)d_in[0];   // (B,N,3) -> b=0, already (N,3)
    const int*   choice = (const int*)  d_in[1];
    const int*   objidx = (const int*)  d_in[2];
    const float* psp    = (const float*)d_in[3];
#define PW(i) ((const float*)d_in[4 + 2 * (i)])
#define PB(i) ((const float*)d_in[4 + 2 * (i) + 1])
    const float* anchors = (const float*)d_in[46];

    // ---- workspace (batch 0 only); (N, C) layouts, f16 activations ----
    char* base = (char*)d_ws;
    size_t off = 0;
    auto alloc = [&](size_t bytes) { void* p = base + off; off += (bytes + 255) & ~(size_t)255; return p; };
    int*       idxb  = (int*)      alloc(sizeof(int) * Nn * KNN_K);
    float*     wpad  = (float*)    alloc(sizeof(float) * 64 * 32);
    _Float16*  x1    = (_Float16*) alloc(2ull * Nn * 64);
    _Float16*  tcat  = (_Float16*) alloc(2ull * Nn * 1152);  // [x_m 0:64 | y_m 64:128 | tpool 128:1152]
    _Float16*  fusion= (_Float16*) alloc(2ull * Nn * 256);   // [x3 0:128 | y2 128:256]
    _Float16*  ygat  = (_Float16*) alloc(2ull * Nn * 32);
    _Float16*  t1    = (_Float16*) alloc(2ull * Nn * 256);
    _Float16*  t2    = (_Float16*) alloc(2ull * Nn * 1024);
    _Float16*  tpool = (_Float16*) alloc(2ull * 1024);
    _Float16*  r1    = (_Float16*) alloc(2ull * Nn * 256);
    _Float16*  r2    = (_Float16*) alloc(2ull * Nn * 1024);
    _Float16*  rpool = (_Float16*) alloc(2ull * 1024);
    _Float16*  t3    = (_Float16*) alloc(2ull * Nn * 512);
    _Float16*  t4    = (_Float16*) alloc(2ull * Nn * 256);
    _Float16*  t5    = (_Float16*) alloc(2ull * Nn * 128);
    float*     t6    = (float*)    alloc(4ull * Nn * 63);
    _Float16*  c1    = (_Float16*) alloc(2ull * 512);
    _Float16*  c2    = (_Float16*) alloc(2ull * 256);
    _Float16*  c3    = (_Float16*) alloc(2ull * 128);
    float*     c4    = (float*)    alloc(4ull * 256);
    _Float16*  rr1   = (_Float16*) alloc(2ull * 512);
    _Float16*  rr2   = (_Float16*) alloc(2ull * 256);
    _Float16*  rr3   = (_Float16*) alloc(2ull * 128);
    float*     rr4   = (float*)    alloc(4ull * 1024);
    (void)ws_size; (void)in_sizes; (void)n_in; (void)out_size;

    auto gemm16 = [&](int pi, const _Float16* X, int ldX, _Float16* Y, int ldY,
                      int O, int Cin, int Np, int act) {
        dim3 g((O + 15) / 16, (Np + 255) / 256);
        gemm_act_kernel<_Float16><<<g, 128, 0, stream>>>(PW(pi), PB(pi), X, Y, O, Cin, Np, ldX, ldY, act);
    };
    auto gemm32 = [&](int pi, const _Float16* X, int ldX, float* Y, int ldY,
                      int O, int Cin, int Np, int act) {
        dim3 g((O + 15) / 16, (Np + 255) / 256);
        gemm_act_kernel<float><<<g, 128, 0, stream>>>(PW(pi), PB(pi), X, Y, O, Cin, Np, ldX, ldY, act);
    };

    // ---- geometry ----
    knn_kernel<<<(Nn + 255) / 256, 256, 0, stream>>>(cloud, idxb, Nn);
    pad_w1_kernel<<<(64 * 32 + 255) / 256, 256, 0, stream>>>(PW(0), wpad);

    // ---- edge convs (WMMA, fused gather + k-max) ----
    edgeconv_kernel<float><<<dim3(4, Nn / 4), 128, 0, stream>>>(
        wpad, PB(0), cloud, idxb, x1, 64, 3, Nn, 3, 64, 32);
    edgeconv_kernel<_Float16><<<dim3(4, Nn / 4), 128, 0, stream>>>(
        PW(1), PB(1), x1, idxb, tcat /*x_m*/, 64, 64, Nn, 64, 1152, 128);
    edgeconv_kernel<_Float16><<<dim3(8, Nn / 4), 128, 0, stream>>>(
        PW(2), PB(2), tcat /*x_m*/, idxb, fusion /*x3*/, 128, 64, Nn, 1152, 256, 128);

    // ---- psp branch ----
    gather_kernel<<<(Nn * PSPC + 255) / 256, 256, 0, stream>>>(psp, choice, ygat, Nn);
    gemm16(3, ygat, 32, tcat + 64 /*y_m*/, 1152, 64, 32, Nn, 1);     // conv1
    gemm16(4, tcat + 64, 1152, fusion + 128 /*y2*/, 256, 128, 64, Nn, 1); // conv2

    // ---- t / r trunks on fusion ----
    gemm16(5,  fusion, 256, t1, 256, 256, 256, Nn, 1);               // conv1_t
    gemm16(6,  t1, 256, t2, 1024, 1024, 256, Nn, 1);                 // conv2_t
    meanpool_kernel<<<4, 256, 0, stream>>>(t2, tpool, 1024, Nn);
    bcast_kernel<<<(Nn * 1024 + 255) / 256, 256, 0, stream>>>(tpool, tcat, Nn);

    gemm16(11, fusion, 256, r1, 256, 256, 256, Nn, 1);               // conv1_r
    gemm16(12, r1, 256, r2, 1024, 1024, 256, Nn, 1);                 // conv2_r
    meanpool_kernel<<<4, 256, 0, stream>>>(r2, rpool, 1024, Nn);

    // ---- translation head ----
    gemm16(7,  tcat, 1152, t3, 512, 512, 1152, Nn, 1);               // conv3_t
    gemm16(8,  t3, 512, t4, 256, 256, 512, Nn, 1);                   // conv4_t
    gemm16(9,  t4, 256, t5, 128, 128, 256, Nn, 1);                   // conv5_t
    gemm32(10, t5, 128, t6, 63, OBJ_N * 3, 128, Nn, 0);              // conv6_t

    // ---- confidence head (pooled) ----
    gemm16(17, rpool, 1024, c1, 512, 512, 1024, 1, 1);               // conv1_c
    gemm16(18, c1, 512, c2, 256, 256, 512, 1, 1);                    // conv2_c
    gemm16(19, c2, 256, c3, 128, 128, 256, 1, 1);                    // conv3_c
    gemm32(20, c3, 128, c4, 252, OBJ_N * ROT_R, 128, 1, 2);          // conv4_c (sigmoid)

    // ---- rotation head (pooled) ----
    gemm16(13, rpool, 1024, rr1, 512, 512, 1024, 1, 1);              // conv3_r
    gemm16(14, rr1, 512, rr2, 256, 256, 512, 1, 1);                  // conv4_r
    gemm16(15, rr2, 256, rr3, 128, 128, 256, 1, 1);                  // conv5_r
    gemm32(16, rr3, 128, rr4, 1008, OBJ_N * ROT_R * 4, 128, 1, 0);   // conv6_r

    // ---- final assembly ----
    head_kernel<<<1, 256, 0, stream>>>(t6, c4, rr4, objidx, anchors, (float*)d_out, Nn);
#undef PW
#undef PB
}
